// ThermoRNN_72249939853609
// MI455X (gfx1250) — matrix operand order repaired and checked by reading, not compiled, not run on hardware
//
#include <hip/hip_runtime.h>
#include <math.h>

#define BN_EPS 1e-5f

typedef __attribute__((ext_vector_type(16))) _Float16 v16h;
typedef __attribute__((ext_vector_type(8)))  float    v8f;

// Broadcast lane `src`'s value of v to all lanes (wave32, uniform src).
__device__ __forceinline__ float lane_bcast(float v, int src) {
    return __builtin_bit_cast(float,
        __builtin_amdgcn_readlane(__builtin_bit_cast(int, v), src));
}

// Native gfx1250 tanh if the builtin exists, else OCML.
__device__ __forceinline__ float fast_tanh(float x) {
#if __has_builtin(__builtin_amdgcn_tanhf)
    return __builtin_amdgcn_tanhf(x);
#elif __has_builtin(__builtin_amdgcn_tanh_f32)
    return __builtin_amdgcn_tanh_f32(x);
#else
    return tanhf(x);
#endif
}

// ---------------------------------------------------------------------------
// WMMA input projection:  dst[L,32] = act(src)[L,K] @ W[30,K]^T + (b_ih+b_hh)
// UseBN=false: act = identity.
// UseBN=true : act = y = max(v*A[k]+B[k], 0), A/B staged in LDS.
// One wave per 16-row M tile. f16 hi/lo split -> ~f32 accuracy, 3 WMMAs/chunk.
// Fragment bounds via index clamping + value masking (branch-free); the store
// epilogue takes an unconditional fast path for full tiles (uniform branch).
// ---------------------------------------------------------------------------
template<bool UseBN>
__global__ __launch_bounds__(32)
void proj_wmma(const float* __restrict__ src, int lda, int K,
               const float* __restrict__ W,
               const float* __restrict__ b_ih, const float* __restrict__ b_hh,
               const float* __restrict__ bnAB,
               float* __restrict__ dst, int L)
{
    __shared__ float sAB[64];
    const int lane  = threadIdx.x;
    if (UseBN) {
        sAB[lane]      = bnAB[lane];
        sAB[32 + lane] = bnAB[32 + lane];
        __syncthreads();
    }

    const int tile  = blockIdx.x;
    const int lrow  = lane & 15;
    const int rhalf = lane >> 4;
    const int row   = tile * 16 + lrow;
    const bool fullTile = (tile * 16 + 16 <= L);   // uniform across the wave
    const int rowc  = (row < L) ? row : (L - 1);
    const float rowm = (row < L) ? 1.0f : 0.0f;
    const int nK = (K + 31) >> 5;
    const float* __restrict__ srow = src + (size_t)rowc * lda;

    for (int nt = 0; nt < 2; ++nt) {
        v8f c = {};
        const int n  = nt * 16 + lrow;
        const int nc = (n < 30) ? n : 29;
        const float nm = (n < 30) ? 1.0f : 0.0f;
        const float* __restrict__ wrow = W + nc * K;

        for (int kc = 0; kc < nK; ++kc) {
            const int k0 = kc << 5;
            v16h ah, al, bh, bl;
            #pragma unroll
            for (int e = 0; e < 16; ++e) {
                // A fragment (16x32 f16 layout, ISA 7.12.2)
                int ka  = k0 + ((e < 8) ? 0 : 16) + (rhalf ? 8 : 0) + (e & 7);
                int kac = (ka < K) ? ka : (K - 1);
                float ma = (ka < K) ? rowm : 0.0f;
                float va = srow[kac];
                if (UseBN)
                    va = fmaxf(fmaf(va, sAB[kac], sAB[32 + kac]), 0.0f);
                va *= ma;
                _Float16 vh = (_Float16)va;
                ah[e] = vh;
                al[e] = (_Float16)(va - (float)vh);

                // B fragment (32x16 f16 layout): B[k][n] = W[n][k]
                int kb  = k0 + ((e < 8) ? 0 : 16) + (rhalf ? 8 : 0) + (e & 7);
                int kbc = (kb < K) ? kb : (K - 1);
                float vb = wrow[kbc] * ((kb < K) ? nm : 0.0f);
                _Float16 wh = (_Float16)vb;
                bh[e] = wh;
                bl[e] = (_Float16)(vb - (float)wh);
            }
            c = __builtin_amdgcn_wmma_f32_16x16x32_f16(false, ah, false, bh,
                                                       (short)0, c, false, false);
            c = __builtin_amdgcn_wmma_f32_16x16x32_f16(false, ah, false, bl,
                                                       (short)0, c, false, false);
            c = __builtin_amdgcn_wmma_f32_16x16x32_f16(false, al, false, bh,
                                                       (short)0, c, false, false);
        }
        const float bias = (b_ih[nc] + b_hh[nc]) * nm;
        float* __restrict__ dcol =
            dst + (size_t)(tile * 16 + (rhalf ? 8 : 0)) * 32 + n;
        if (fullTile) {
            #pragma unroll
            for (int r = 0; r < 8; ++r)
                dcol[(size_t)r * 32] = c[r] + bias;
        } else {
            #pragma unroll
            for (int r = 0; r < 8; ++r) {
                int m = tile * 16 + r + (rhalf ? 8 : 0);
                if (m < L) dst[(size_t)m * 32 + n] = c[r] + bias;
            }
        }
    }
}

// ---------------------------------------------------------------------------
// One serial RNN step for this wave: lane i owns hidden unit i.
// readlane only ever touches lanes 0..29, so lanes 30/31 need no masking.
// ---------------------------------------------------------------------------
__device__ __forceinline__ float rnn_step(const float (&w)[30], float h, float xin) {
    float a0 = xin, a1 = 0.0f, a2 = 0.0f, a3 = 0.0f;
    #pragma unroll
    for (int j = 0; j < 28; j += 4) {
        a0 = fmaf(w[j + 0], lane_bcast(h, j + 0), a0);
        a1 = fmaf(w[j + 1], lane_bcast(h, j + 1), a1);
        a2 = fmaf(w[j + 2], lane_bcast(h, j + 2), a2);
        a3 = fmaf(w[j + 3], lane_bcast(h, j + 3), a3);
    }
    a0 = fmaf(w[28], lane_bcast(h, 28), a0);
    a1 = fmaf(w[29], lane_bcast(h, 29), a1);
    return fast_tanh((a0 + a1) + (a2 + a3));
}

// ---------------------------------------------------------------------------
// Serial RNN scan, single wave32. W_hh row i in 30 VGPRs of lane i; xp
// prefetched 8 steps ahead (register ring, clamped addresses, no predication)
// plus a speculative global_prefetch a further chunk out. Epilogue folds
// BatchNorm into per-channel affine A/B (y = h*A + B).
// ---------------------------------------------------------------------------
__global__ __launch_bounds__(32)
void rnn_scan(const float* __restrict__ xp, float* __restrict__ hs,
              const float* __restrict__ W_hh,
              const float* __restrict__ gamma, const float* __restrict__ beta,
              float* __restrict__ bnAB, int L)
{
    const int lane = threadIdx.x;
    const int lc   = (lane < 30) ? lane : 29;

    float w[30];
    #pragma unroll
    for (int j = 0; j < 30; ++j)
        w[j] = (lane < 30) ? W_hh[lane * 30 + j] : 0.0f;

    float h = 0.0f, s1 = 0.0f, s2 = 0.0f;

    const int Lmain = L & ~7;
    float buf[8];
    #pragma unroll
    for (int k = 0; k < 8; ++k) {
        int tp = (k < L) ? k : (L - 1);
        buf[k] = xp[(size_t)tp * 32 + lane];
    }

    int t = 0;
    for (; t < Lmain; t += 8) {
        // speculative prefetch two chunks out (drop-safe if OOB)
        __builtin_prefetch(&xp[(size_t)(t + 16) * 32 + lane], 0, 0);
        // register-ring prefetch of next chunk (clamped, branch-free)
        float nbuf[8];
        #pragma unroll
        for (int k = 0; k < 8; ++k) {
            int tp = t + 8 + k;
            tp = (tp < L) ? tp : (L - 1);
            nbuf[k] = xp[(size_t)tp * 32 + lane];
        }
        #pragma unroll
        for (int k = 0; k < 8; ++k) {
            h = rnn_step(w, h, buf[k]);
            s1 += h;
            s2 = fmaf(h, h, s2);
            hs[(size_t)(t + k) * 32 + lane] = h;
        }
        #pragma unroll
        for (int k = 0; k < 8; ++k) buf[k] = nbuf[k];
    }
    for (; t < L; ++t) {
        h = rnn_step(w, h, xp[(size_t)t * 32 + lane]);
        s1 += h;
        s2 = fmaf(h, h, s2);
        hs[(size_t)t * 32 + lane] = h;
    }

    // Fold training-mode BatchNorm into per-channel affine: y = h*A + B.
    // (Entries for lanes >= 30 are never consumed downstream.)
    const float invLf = 1.0f / (float)L;
    const float mean = s1 * invLf;
    const float var  = fmaf(-mean, mean, s2 * invLf);
    const float A = gamma[lc] * rsqrtf(var + BN_EPS);
    const float B = fmaf(-mean, A, beta[lc]);
    bnAB[lane]      = A;
    bnAB[32 + lane] = B;
}

// ---------------------------------------------------------------------------
// Head: out[t] = relu(h2[t,:]*A + B) @ W_out^T + b_out
// ---------------------------------------------------------------------------
__global__ void head_kernel(const float* __restrict__ hs,
                            const float* __restrict__ bnAB,
                            const float* __restrict__ W_out,
                            const float* __restrict__ b_out,
                            float* __restrict__ out, int L)
{
    const int t = blockIdx.x * blockDim.x + threadIdx.x;
    if (t >= L) return;
    float acc = b_out[0];
    const float* __restrict__ hrow = hs + (size_t)t * 32;
    #pragma unroll
    for (int n = 0; n < 30; ++n) {
        float y = fmaxf(fmaf(hrow[n], bnAB[n], bnAB[32 + n]), 0.0f);
        acc = fmaf(y, W_out[n], acc);
    }
    out[t] = acc;
}

// ---------------------------------------------------------------------------
extern "C" void kernel_launch(void* const* d_in, const int* in_sizes, int n_in,
                              void* d_out, int out_size, void* d_ws, size_t ws_size,
                              hipStream_t stream)
{
    (void)n_in; (void)out_size; (void)ws_size;
    const float* x     = (const float*)d_in[0];
    const float* W_ih1 = (const float*)d_in[1];
    const float* b_ih1 = (const float*)d_in[2];
    const float* W_hh1 = (const float*)d_in[3];
    const float* b_hh1 = (const float*)d_in[4];
    const float* g1    = (const float*)d_in[5];
    const float* be1   = (const float*)d_in[6];
    const float* W_ih2 = (const float*)d_in[7];
    const float* b_ih2 = (const float*)d_in[8];
    const float* W_hh2 = (const float*)d_in[9];
    const float* b_hh2 = (const float*)d_in[10];
    const float* g2    = (const float*)d_in[11];
    const float* be2   = (const float*)d_in[12];
    const float* W_out = (const float*)d_in[13];
    const float* b_out = (const float*)d_in[14];
    float* out = (float*)d_out;

    const int L = in_sizes[0] / 35;

    float* P   = (float*)d_ws;           // L x 32 (xp buffer)
    float* Q   = P + (size_t)L * 32;     // L x 32 (hidden-state buffer)
    float* ab1 = Q + (size_t)L * 32;     // 64 floats: BN affine A|B (layer 1)
    float* ab2 = ab1 + 64;               // 64 floats: BN affine A|B (layer 2)

    const dim3 gGemm((L + 15) / 16), bWave(32);

    // Layer 1: P = x @ W_ih1^T + (b_ih1 + b_hh1)   [WMMA]
    proj_wmma<false><<<gGemm, bWave, 0, stream>>>(x, 35, 35, W_ih1, b_ih1,
                                                  b_hh1, nullptr, P, L);
    // Layer 1 scan: Q = h1 sequence, ab1 = fused BN affine
    rnn_scan<<<1, bWave, 0, stream>>>(P, Q, W_hh1, g1, be1, ab1, L);
    // Layer 2: P = relu(Q*A+B) @ W_ih2^T + (b_ih2 + b_hh2)   [WMMA, BN in LDS]
    proj_wmma<true><<<gGemm, bWave, 0, stream>>>(Q, 32, 30, W_ih2, b_ih2,
                                                 b_hh2, ab1, P, L);
    // Layer 2 scan
    rnn_scan<<<1, bWave, 0, stream>>>(P, Q, W_hh2, g2, be2, ab2, L);
    // Head
    head_kernel<<<(L + 255) / 256, 256, 0, stream>>>(Q, ab2, W_out, b_out,
                                                     out, L);
}
